// MultiHeadAttention_49263274885339
// MI455X (gfx1250) — matrix operand order repaired and checked by reading, not compiled, not run on hardware
//
#include <hip/hip_runtime.h>
#include <hip/hip_bf16.h>
#include <stdint.h>

#define B_  8
#define S_  1024
#define D_  1024
#define H_  16
#define DK_ 64
#define DV_ 64

typedef __bf16 v16bf __attribute__((ext_vector_type(16)));
typedef float  v8f   __attribute__((ext_vector_type(8)));
typedef unsigned int v4u __attribute__((ext_vector_type(4)));

// 32-byte matrix fragment: 16 bf16 values per lane (A or B operand of
// v_wmma_f32_16x16x32_bf16), overlaid with 2x16B for b128 LDS loads.
union Frag {
    v4u   q[2];
    v16bf v;
};

__device__ __forceinline__ unsigned short f2bf(float f) {
    unsigned int u = __float_as_uint(f);
    u += 0x7fffu + ((u >> 16) & 1u);   // round-to-nearest-even
    return (unsigned short)(u >> 16);
}

// CDNA5 async global->LDS copy, 16 bytes per lane. Tracked by ASYNCcnt.
// LDS destination operand is the per-lane LDS byte offset (= low 32 bits of
// the generic shared-memory address, per the aperture mapping).
__device__ __forceinline__ void async_copy_b128(const unsigned short* gptr,
                                                unsigned short* lptr) {
    unsigned lds_off = (unsigned)(uintptr_t)lptr;
    unsigned long long ga = (unsigned long long)(uintptr_t)gptr;
    asm volatile("global_load_async_to_lds_b128 %0, %1, off"
                 :: "v"(lds_off), "v"(ga) : "memory");
}
__device__ __forceinline__ void wait_async() {
    asm volatile("s_wait_asynccnt 0x0" ::: "memory");
}

// ---------------------------------------------------------------------------
// f32 -> bf16 conversion (straight copy)
// ---------------------------------------------------------------------------
__global__ void cvt_bf16_kernel(const float* __restrict__ src,
                                unsigned short* __restrict__ dst, int n) {
    int i = blockIdx.x * blockDim.x + threadIdx.x;
    if (i < n) dst[i] = f2bf(src[i]);
}

// f32 (KxN) -> bf16 transposed (NxK), so GEMM B-tiles are contiguous in K.
__global__ void cvt_wt_kernel(const float* __restrict__ src,
                              unsigned short* __restrict__ dst, int N, int K) {
    int i = blockIdx.x * blockDim.x + threadIdx.x;
    if (i < N * K) {
        int n = i / K, k = i - n * K;
        dst[i] = f2bf(src[(size_t)k * N + n]);
    }
}

// ---------------------------------------------------------------------------
// Tiled bf16 GEMM:  out = X(MxK) @ Wt^T + bias     (Wt is N x K row-major)
//   mode 0: f32 output, row-major MxN              (output projection)
//   mode 1: bf16 output scattered to [B,H,S,64]    (Q/K projection), *scale
//   mode 2: bf16 output scattered to [B,H,64,S]    (V projection, transposed)
// Block: 256 threads (8 waves). C tile 128(M) x 128(N); each wave owns a
// 32x64 subtile (2 A-frags x 4 B-frags -> 8 WMMA, B reused across A rows).
// K stepped by 32, double-buffered through LDS with async global->LDS copies.
// ---------------------------------------------------------------------------
__global__ void __launch_bounds__(256)
gemm_bf16_kernel(const unsigned short* __restrict__ X,
                 const unsigned short* __restrict__ Wt,
                 const float* __restrict__ bias,
                 float* __restrict__ outf,
                 unsigned short* __restrict__ outq,
                 int M, int N, int K, int mode, float scale) {
    __shared__ unsigned short lds_x[2][128 * 32];   // X tile [m][k]
    __shared__ unsigned short lds_w[2][128 * 32];   // W tile [n][k]

    const int tid  = threadIdx.x;
    const int lane = tid & 31;
    const int wv   = tid >> 5;
    const int m0   = blockIdx.y * 128;
    const int n0   = blockIdx.x * 128;
    const int mt   = wv & 3;        // M subtile (32 rows) of this wave
    const int nh   = wv >> 2;       // N half (64 cols) of this wave
    const int lrow = lane & 15;
    const int hi   = lane >> 4;     // lane half: selects K sub-ranges

    auto stage = [&](int buf, int k0) {
#pragma unroll
        for (int j = 0; j < 2; ++j) {               // X: 128x32 = 512 x b128
            int i = tid + j * 256;
            int row = i >> 2, g = i & 3;
            async_copy_b128(X + (size_t)(m0 + row) * K + k0 + g * 8,
                            &lds_x[buf][row * 32 + g * 8]);
        }
#pragma unroll
        for (int j = 0; j < 2; ++j) {               // W: 128x32 = 512 x b128
            int i = tid + j * 256;
            int n = i >> 2, g = i & 3;
            async_copy_b128(Wt + (size_t)(n0 + n) * K + k0 + g * 8,
                            &lds_w[buf][n * 32 + g * 8]);
        }
    };

    v8f acc[2][4];
#pragma unroll
    for (int u = 0; u < 2; ++u)
#pragma unroll
        for (int t = 0; t < 4; ++t) acc[u][t] = {};

    stage(0, 0);
    wait_async();
    __syncthreads();

    int cur = 0;
    for (int k0 = 0; k0 < K; k0 += 32) {
        if (k0 + 32 < K) stage(cur ^ 1, k0 + 32);

        // A fragments (16x32): lane<16 -> K {0..7,16..23}, lane>=16 -> {8..15,24..31}
        Frag a[2];
#pragma unroll
        for (int u = 0; u < 2; ++u) {
            const v4u* p = (const v4u*)&lds_x[cur][(mt * 32 + u * 16 + lrow) * 32 + hi * 8];
            a[u].q[0] = p[0];
            a[u].q[1] = p[2];
        }
#pragma unroll
        for (int t = 0; t < 4; ++t) {
            // B fragment (32x16): lane = column n, K {0..7,16..23} / {8..15,24..31}
            Frag b;
            const v4u* p = (const v4u*)&lds_w[cur][(nh * 64 + t * 16 + lrow) * 32 + hi * 8];
            b.q[0] = p[0];
            b.q[1] = p[2];
#pragma unroll
            for (int u = 0; u < 2; ++u)
                acc[u][t] = __builtin_amdgcn_wmma_f32_16x16x32_bf16(
                    false, a[u].v, false, b.v, (short)0, acc[u][t], false, false);
        }
        wait_async();
        __syncthreads();
        cur ^= 1;
    }

    // Store C: VGPR v -> row, lane -> column
#pragma unroll
    for (int t = 0; t < 4; ++t) {
        int col = n0 + nh * 64 + t * 16 + lrow;
        float bc = bias[col];
#pragma unroll
        for (int u = 0; u < 2; ++u) {
#pragma unroll
            for (int v = 0; v < 8; ++v) {
                int row = m0 + mt * 32 + u * 16 + v + hi * 8;
                float val = (acc[u][t][v] + bc) * scale;
                if (mode == 0) {
                    outf[(size_t)row * N + col] = val;
                } else {
                    int b = row >> 10, s = row & 1023;       // row = b*S + s
                    int h = col >> 6,  dk = col & 63;        // col = h*64 + dk
                    if (mode == 1) {
                        outq[(((size_t)(b * H_ + h)) * S_ + s) * DK_ + dk] = f2bf(val);
                    } else {   // V: [B,H,DV,S] so attention can stage V^T directly
                        outq[(((size_t)(b * H_ + h)) * DV_ + dk) * S_ + s] = f2bf(val);
                    }
                }
            }
        }
    }
}

// ---------------------------------------------------------------------------
// Flash attention over one (b, h, 64-q-row) block. 128 threads = 4 waves,
// each wave owns a 16-row q tile. Keys streamed in 32-key chunks, double-
// buffered in LDS via async copies. Q is pre-scaled by 1/sqrt(DK).
// ---------------------------------------------------------------------------
__global__ void __launch_bounds__(128)
attn_kernel(const unsigned short* __restrict__ Q,
            const unsigned short* __restrict__ Kb,
            const unsigned short* __restrict__ Vt,
            const int* __restrict__ masks,
            unsigned short* __restrict__ AO) {
    __shared__ unsigned short lds_k[2][32 * 64];     // K chunk [key][dk]
    __shared__ unsigned short lds_vt[2][64 * 32];    // V^T chunk [dv][key]
    __shared__ unsigned short lds_p[4 * 16 * 32];    // per-wave P tile [q][key]
    __shared__ int lds_mask[2][32];

    const int tid  = threadIdx.x;
    const int lane = tid & 31;
    const int wv   = tid >> 5;
    const int lrow = lane & 15;
    const int hi   = lane >> 4;

    const int blk = blockIdx.x;              // B*H*(S/64) = 2048 blocks
    const int qb  = blk & 15;
    const int h   = (blk >> 4) & 15;
    const int b   = blk >> 8;
    const int q0  = qb * 64 + wv * 16;

    const size_t head = ((size_t)(b * H_ + h)) * S_ * DK_;
    const unsigned short* qbase = Q  + head;
    const unsigned short* kbase = Kb + head;
    const unsigned short* vtbase = Vt + head;    // [dv][s] within head

    auto stage = [&](int buf, int kc) {
#pragma unroll
        for (int j = 0; j < 2; ++j) {            // K: 32x64 = 256 x b128
            int i = tid + j * 128;
            int key = i >> 3, g = i & 7;
            async_copy_b128(kbase + (size_t)(kc + key) * DK_ + g * 8,
                            &lds_k[buf][key * 64 + g * 8]);
        }
#pragma unroll
        for (int j = 0; j < 2; ++j) {            // V^T: 64x32 = 256 x b128
            int i = tid + j * 128;
            int dv = i >> 2, g = i & 3;
            async_copy_b128(vtbase + (size_t)dv * S_ + kc + g * 8,
                            &lds_vt[buf][dv * 32 + g * 8]);
        }
        if (tid < 32) lds_mask[buf][tid] = masks[(size_t)b * S_ + kc + tid];
    };

    // Q fragments for both 32-wide halves of DK, kept in registers
    Frag aq[2];
#pragma unroll
    for (int kd = 0; kd < 2; ++kd) {
        const v4u* p = (const v4u*)(qbase + (size_t)(q0 + lrow) * DK_ + kd * 32 + hi * 8);
        aq[kd].q[0] = p[0];
        aq[kd].q[1] = p[2];
    }

    v8f o[4];
#pragma unroll
    for (int t = 0; t < 4; ++t) o[t] = {};
    float mrow[8], lsum[8];
#pragma unroll
    for (int v = 0; v < 8; ++v) { mrow[v] = -1e30f; lsum[v] = 0.f; }

    stage(0, 0);
    wait_async();
    __syncthreads();

    int cur = 0;
    for (int kc = 0; kc < S_; kc += 32) {
        if (kc + 32 < S_) stage(cur ^ 1, kc + 32);

        // Logits: 16 q-rows x 32 keys = two 16x16 WMMA tiles, K-dim = 64
        v8f s[2];
#pragma unroll
        for (int nt = 0; nt < 2; ++nt) {
            s[nt] = {};
#pragma unroll
            for (int kd = 0; kd < 2; ++kd) {
                Frag bk;  // column = key, contraction = dk
                const v4u* p = (const v4u*)&lds_k[cur][(nt * 16 + lrow) * 64 + kd * 32 + hi * 8];
                bk.q[0] = p[0];
                bk.q[1] = p[2];
                s[nt] = __builtin_amdgcn_wmma_f32_16x16x32_bf16(
                    false, aq[kd].v, false, bk.v, (short)0, s[nt], false, false);
            }
            if (lds_mask[cur][nt * 16 + lrow] == 0) {
#pragma unroll
                for (int v = 0; v < 8; ++v) s[nt][v] = -1.0e9f;
            }
        }

        // Online softmax; VGPR index v holds row (v + hi*8), cols across 16 lanes
#pragma unroll
        for (int v = 0; v < 8; ++v) {
            float rm = fmaxf(s[0][v], s[1][v]);
#pragma unroll
            for (int msk = 1; msk < 16; msk <<= 1)
                rm = fmaxf(rm, __shfl_xor(rm, msk, 32));
            float mnew  = fmaxf(mrow[v], rm);
            float alpha = __expf(mrow[v] - mnew);
            float p0 = __expf(s[0][v] - mnew);
            float p1 = __expf(s[1][v] - mnew);
            float rs = p0 + p1;
#pragma unroll
            for (int msk = 1; msk < 16; msk <<= 1)
                rs += __shfl_xor(rs, msk, 32);
            lsum[v] = lsum[v] * alpha + rs;
            mrow[v] = mnew;
#pragma unroll
            for (int t = 0; t < 4; ++t) o[t][v] *= alpha;
            // Re-layout P via LDS: C layout -> A-fragment layout
            int prow = v + hi * 8;
            lds_p[wv * 512 + prow * 32 + lrow]      = f2bf(p0);
            lds_p[wv * 512 + prow * 32 + 16 + lrow] = f2bf(p1);
        }

        // P(16x32) @ V(32x64): 4 WMMA over dv tiles
        Frag ap;
        {
            const v4u* p = (const v4u*)&lds_p[wv * 512 + lrow * 32 + hi * 8];
            ap.q[0] = p[0];
            ap.q[1] = p[2];
        }
#pragma unroll
        for (int t = 0; t < 4; ++t) {
            Frag bv;
            const v4u* p = (const v4u*)&lds_vt[cur][(t * 16 + lrow) * 32 + hi * 8];
            bv.q[0] = p[0];
            bv.q[1] = p[2];
            o[t] = __builtin_amdgcn_wmma_f32_16x16x32_bf16(
                false, ap.v, false, bv.v, (short)0, o[t], false, false);
        }
        wait_async();
        __syncthreads();
        cur ^= 1;
    }

    // Normalize and store to [B, S, H*DV] bf16
#pragma unroll
    for (int v = 0; v < 8; ++v) {
        float inv = 1.0f / lsum[v];
        int row = q0 + v + hi * 8;
#pragma unroll
        for (int t = 0; t < 4; ++t) {
            int dv = t * 16 + lrow;
            AO[((size_t)b * S_ + row) * (H_ * DV_) + h * DV_ + dv] = f2bf(o[t][v] * inv);
        }
    }
}

// ---------------------------------------------------------------------------
extern "C" void kernel_launch(void* const* d_in, const int* in_sizes, int n_in,
                              void* d_out, int out_size, void* d_ws, size_t ws_size,
                              hipStream_t stream) {
    const float* x     = (const float*)d_in[0];
    const int*   masks = (const int*)d_in[1];
    const float* Wq    = (const float*)d_in[2];
    const float* bq    = (const float*)d_in[3];
    const float* Wk    = (const float*)d_in[4];
    const float* bk    = (const float*)d_in[5];
    const float* Wv    = (const float*)d_in[6];
    const float* bv    = (const float*)d_in[7];
    const float* Wo    = (const float*)d_in[8];
    const float* bo    = (const float*)d_in[9];

    const size_t NX = (size_t)B_ * S_ * D_;   // 8,388,608
    const size_t NW = (size_t)D_ * D_;        // 1,048,576

    unsigned short* ws  = (unsigned short*)d_ws;
    unsigned short* xb  = ws;
    unsigned short* wqb = xb  + NX;   // W^T (N x K) bf16
    unsigned short* wkb = wqb + NW;
    unsigned short* wvb = wkb + NW;
    unsigned short* wob = wvb + NW;
    unsigned short* qb  = wob + NW;   // [B,H,S,DK] bf16
    unsigned short* kb  = qb  + NX;
    unsigned short* vtb = kb  + NX;   // [B,H,DV,S] bf16
    unsigned short* aob = vtb + NX;   // [B,S,H*DV] bf16

    // f32 -> bf16 (weights transposed for K-contiguous B tiles)
    cvt_bf16_kernel<<<(int)((NX + 255) / 256), 256, 0, stream>>>(x, xb, (int)NX);
    cvt_wt_kernel<<<(int)((NW + 255) / 256), 256, 0, stream>>>(Wq, wqb, D_, D_);
    cvt_wt_kernel<<<(int)((NW + 255) / 256), 256, 0, stream>>>(Wk, wkb, D_, D_);
    cvt_wt_kernel<<<(int)((NW + 255) / 256), 256, 0, stream>>>(Wv, wvb, D_, D_);
    cvt_wt_kernel<<<(int)((NW + 255) / 256), 256, 0, stream>>>(Wo, wob, D_, D_);

    const int M = B_ * S_, N = D_, K = D_;
    dim3 ggrid(N / 128, M / 128);   // (8, 64)

    // Q/K/V projections (Q folded with 1/sqrt(DK); V stored transposed)
    gemm_bf16_kernel<<<ggrid, 256, 0, stream>>>(xb, wqb, bq, nullptr, qb,  M, N, K, 1, 0.125f);
    gemm_bf16_kernel<<<ggrid, 256, 0, stream>>>(xb, wkb, bk, nullptr, kb,  M, N, K, 1, 1.0f);
    gemm_bf16_kernel<<<ggrid, 256, 0, stream>>>(xb, wvb, bv, nullptr, vtb, M, N, K, 2, 1.0f);

    // Fused flash attention
    attn_kernel<<<B_ * H_ * (S_ / 64), 128, 0, stream>>>(qb, kb, vtb, masks, aob);

    // Output projection -> f32 d_out
    gemm_bf16_kernel<<<ggrid, 256, 0, stream>>>(aob, wob, bo, (float*)d_out, nullptr, M, N, K, 0, 1.0f);
}
